// LossFunc_27470610825634
// MI455X (gfx1250) — matrix operand order repaired and checked by reading, not compiled, not run on hardware
//
#include <hip/hip_runtime.h>

typedef _Float16 v4h  __attribute__((ext_vector_type(4)));
typedef _Float16 v8h  __attribute__((ext_vector_type(8)));
typedef _Float16 v16h __attribute__((ext_vector_type(16)));
typedef float    v8f  __attribute__((ext_vector_type(8)));

#define NPTS 512
#define DIMS 128
#define NMC  64
#define MARGIN 2.0f

static constexpr size_t X_ELEMS    = (size_t)3 * NMC * NPTS * DIMS;   // f16 elems
static constexpr size_t NORM_ELEMS = (size_t)3 * NMC * NPTS;          // f32 elems

// ---------------------------------------------------------------------------
// Kernel 1: X = mu + sigma*eps  (f32 -> f16), plus row norms of rounded X.
// One wave (32 lanes) per 128-element row; lane handles 4 consecutive floats.
// ---------------------------------------------------------------------------
__global__ void build_x_kernel(const float* __restrict__ mu,
                               const float* __restrict__ sigma,
                               const float* __restrict__ epsA,
                               const float* __restrict__ epsB,
                               const float* __restrict__ epsC,
                               _Float16* __restrict__ X,
                               float* __restrict__ norms) {
    const int wave = threadIdx.x >> 5;
    const int lane = threadIdx.x & 31;
    const int row  = blockIdx.x * 8 + wave;          // [0, 3*NMC*NPTS)
    const int g    = row / (NMC * NPTS);             // 0=A,1=B,2=C
    const int rem  = row - g * (NMC * NPTS);
    const int m    = rem / NPTS;
    const int n    = rem - m * NPTS;

    const float* __restrict__ eps = (g == 0) ? epsA : (g == 1) ? epsB : epsC;
    const size_t muOff  = ((size_t)(g * NPTS + n)) * DIMS + lane * 4;
    const size_t epsOff = (((size_t)m * NPTS + n)) * DIMS + lane * 4;

    const float4 m4 = *(const float4*)(mu + muOff);
    const float4 s4 = *(const float4*)(sigma + muOff);
    const float4 e4 = *(const float4*)(eps + epsOff);

    const float x0 = m4.x + s4.x * e4.x;
    const float x1 = m4.y + s4.y * e4.y;
    const float x2 = m4.z + s4.z * e4.z;
    const float x3 = m4.w + s4.w * e4.w;

    v4h h = { (_Float16)x0, (_Float16)x1, (_Float16)x2, (_Float16)x3 };
    *(v4h*)(X + (size_t)row * DIMS + lane * 4) = h;

    // norm from the rounded values so d2 = |x|^2+|y|^2-2x.y stays consistent
    const float r0 = (float)h[0], r1 = (float)h[1], r2 = (float)h[2], r3 = (float)h[3];
    float s = r0 * r0 + r1 * r1 + r2 * r2 + r3 * r3;
    for (int off = 16; off > 0; off >>= 1) s += __shfl_down(s, off, 32);
    if (lane == 0) norms[row] = s;
}

// ---------------------------------------------------------------------------
// WMMA fragment loaders (NT addressing; both operands read row-major).
// A 16x32 f16 per-lane layout: rows = lane&15; K = {sel..sel+7, sel+16..sel+23},
//   sel = (lane>>4)*8  -> two 16B chunks at base, base+16.
// B 32x16 f16 per-lane layout: col = lane&15; K = {sel..sel+7, sel+16..sel+23},
//   sel = (lane>>4)*8  -> two 16B chunks at base, base+16.
// ---------------------------------------------------------------------------
__device__ __forceinline__ v16h load_frag_a(const _Float16* base) {
    v8h c0 = *(const v8h*)(base);
    v8h c1 = *(const v8h*)(base + 16);
    return __builtin_shufflevector(c0, c1, 0,1,2,3,4,5,6,7,8,9,10,11,12,13,14,15);
}
__device__ __forceinline__ v16h load_frag_b(const _Float16* base) {
    v8h c0 = *(const v8h*)(base);
    v8h c1 = *(const v8h*)(base + 16);
    return __builtin_shufflevector(c0, c1, 0,1,2,3,4,5,6,7,8,9,10,11,12,13,14,15);
}

// ---------------------------------------------------------------------------
// Kernel 2: one block = one 128x128 output tile of one (pair, m) distance
// matrix. 8 waves; each wave owns a 32x64 sub-tile (2x4 WMMA accumulators).
// K=128 in 4 steps of v_wmma_f32_16x16x32_f16. Epilogue folds the loss
// transform and reduces to a single double accumulator.
// ---------------------------------------------------------------------------
__global__ void pair_tiles_kernel(const _Float16* __restrict__ X,
                                  const float* __restrict__ norms,
                                  double* __restrict__ acc) {
    const int tile = blockIdx.x;   // 0..15 : 4x4 tiling of 512x512
    const int m    = blockIdx.y;   // 0..63
    const int p    = blockIdx.z;   // 0: (A,C) pos, 1: (B,A) neg, 2: (B,C) neg

    const int gl = (p == 0) ? 0 : 1;
    const int gr = (p == 1) ? 0 : 2;

    const _Float16* __restrict__ XL = X + ((size_t)gl * NMC + m) * NPTS * DIMS;
    const _Float16* __restrict__ XR = X + ((size_t)gr * NMC + m) * NPTS * DIMS;
    const float* __restrict__ nL = norms + ((size_t)gl * NMC + m) * NPTS;
    const float* __restrict__ nR = norms + ((size_t)gr * NMC + m) * NPTS;

    const int tileRow = (tile >> 2) * 128;
    const int tileCol = (tile & 3) * 128;
    const int wave = threadIdx.x >> 5;
    const int lane = threadIdx.x & 31;
    const int waveRow = tileRow + (wave >> 1) * 32;   // 4 row groups of 32
    const int waveCol = tileCol + (wave & 1) * 64;    // 2 col groups of 64

    const int frow = lane & 15;
    const int aSel = (lane >> 4) * 8;
    const int bSel = (lane >> 4) * 8;

    v8f c[2][4] = {};

    #pragma unroll
    for (int k0 = 0; k0 < DIMS; k0 += 32) {
        v16h a[2], b[4];
        #pragma unroll
        for (int tm = 0; tm < 2; ++tm)
            a[tm] = load_frag_a(XL + (size_t)(waveRow + tm * 16 + frow) * DIMS + k0 + aSel);
        #pragma unroll
        for (int tn = 0; tn < 4; ++tn)
            b[tn] = load_frag_b(XR + (size_t)(waveCol + tn * 16 + frow) * DIMS + k0 + bSel);
        #pragma unroll
        for (int tm = 0; tm < 2; ++tm)
            #pragma unroll
            for (int tn = 0; tn < 4; ++tn)
                c[tm][tn] = __builtin_amdgcn_wmma_f32_16x16x32_f16(
                    false, a[tm], false, b[tn], (short)0, c[tm][tn], false, false);
    }

    // Epilogue: d2 = |x|^2 + |y|^2 - 2 x.y ; pos: clamp(d2); neg: relu(M-d)^2
    float ls = 0.0f;
    #pragma unroll
    for (int tm = 0; tm < 2; ++tm) {
        #pragma unroll
        for (int tn = 0; tn < 4; ++tn) {
            const int col = waveCol + tn * 16 + (lane & 15);
            const float ny = nR[col];
            #pragma unroll
            for (int e = 0; e < 8; ++e) {
                const int r = waveRow + tm * 16 + (lane >> 4) * 8 + e;
                float d2 = nL[r] + ny - 2.0f * c[tm][tn][e];
                d2 = fmaxf(d2, 1e-12f);
                if (p == 0) {
                    ls += d2;
                } else {
                    const float d = sqrtf(d2);
                    const float t = fmaxf(MARGIN - d, 0.0f);
                    ls += t * t;
                }
            }
        }
    }

    for (int off = 16; off > 0; off >>= 1) ls += __shfl_down(ls, off, 32);
    __shared__ float red[8];
    if (lane == 0) red[wave] = ls;
    __syncthreads();
    if (threadIdx.x == 0) {
        float s = 0.0f;
        #pragma unroll
        for (int i = 0; i < 8; ++i) s += red[i];
        atomicAdd(acc, (double)s);
    }
}

__global__ void zero_acc_kernel(double* acc) { acc[0] = 0.0; }

__global__ void finalize_kernel(const double* __restrict__ acc, float* __restrict__ out) {
    // all three means share divisor NMC*NPTS*NPTS = 16777216
    out[0] = (float)(acc[0] * (1.0 / 16777216.0));
}

extern "C" void kernel_launch(void* const* d_in, const int* in_sizes, int n_in,
                              void* d_out, int out_size, void* d_ws, size_t ws_size,
                              hipStream_t stream) {
    const float* mu    = (const float*)d_in[0];
    const float* sigma = (const float*)d_in[1];
    const float* epsA  = (const float*)d_in[2];
    const float* epsB  = (const float*)d_in[3];
    const float* epsC  = (const float*)d_in[4];
    float* out = (float*)d_out;

    char* ws = (char*)d_ws;
    _Float16* X  = (_Float16*)ws;
    float* norms = (float*)(ws + X_ELEMS * sizeof(_Float16));
    double* acc  = (double*)(ws + X_ELEMS * sizeof(_Float16) + NORM_ELEMS * sizeof(float));

    zero_acc_kernel<<<1, 1, 0, stream>>>(acc);

    // 3*64*512 rows, 8 rows (waves) per 256-thread block
    build_x_kernel<<<(3 * NMC * NPTS) / 8, 256, 0, stream>>>(mu, sigma, epsA, epsB, epsC, X, norms);

    dim3 grid2(16, NMC, 3);
    pair_tiles_kernel<<<grid2, 256, 0, stream>>>(X, norms, acc);

    finalize_kernel<<<1, 1, 0, stream>>>(acc, out);
}
